// RSSM_62680752718523
// MI455X (gfx1250) — matrix operand order repaired and checked by reading, not compiled, not run on hardware
//
#include <hip/hip_runtime.h>

// ---------------------------------------------------------------------------
// RSSM observe rollout for MI455X (gfx1250).
// All seven GEMMs per step run on V_WMMA_F32_16X16X32_BF16 (wave32 matrix
// pipe), f32 accumulate. Weights converted once per call to bf16 (L2-resident,
// ~8 MB). Sequential T=64 loop -> multi-launch on `stream` (graph-friendly).
// Workspace use: ~21 MB.
// ---------------------------------------------------------------------------

typedef __bf16 bf16;
typedef bf16  bf16x16 __attribute__((ext_vector_type(16)));
typedef float floatx8 __attribute__((ext_vector_type(8)));

#define BB   512      // batch
#define TT   64       // time steps
#define EMB  512
#define DET  512
#define SF   1024     // STOCH*CLS
#define HID  512
#define KPRE 1056     // 1027 padded to multiple of 32
#define OUTW 3584     // DET + SF + SF + SF

// ---------------- WMMA fragment loads (gfx1250 documented layouts) ----------

__device__ __forceinline__ bf16x16 load_frag_a(const bf16* __restrict__ A,
                                               int lda, int m, int k0, int half) {
    // A is MxK row-major. Lane holds row m = base + (lane&15).
    // lanes 0-15: K = {k0..k0+7} u {k0+16..k0+23}; lanes 16-31: +8 shift.
    union { uint4 q[2]; bf16x16 v; } f;
    const bf16* row = A + (size_t)m * lda + k0 + half * 8;
    f.q[0] = *(const uint4*)(row);        // 8 bf16: K = k0+8h .. k0+8h+7
    f.q[1] = *(const uint4*)(row + 16);   // 8 bf16: K = k0+16+8h .. +7
    return f.v;
}

__device__ __forceinline__ bf16x16 load_frag_b(const bf16* __restrict__ W,
                                               int ldb, int n, int k0, int half) {
    // W is NxK row-major (so B[k][n] = W[n][k]). Lane holds col n = base+(lane&15).
    // lanes 0-15: K = {k0..k0+7} u {k0+16..k0+23}; lanes 16-31: +8 shift.
    union { uint4 q[2]; bf16x16 v; } f;
    const bf16* row = W + (size_t)n * ldb + k0 + half * 8;
    f.q[0] = *(const uint4*)(row);
    f.q[1] = *(const uint4*)(row + 16);
    return f.v;
}

// ---------------- generic bf16 WMMA GEMM:  C = A * W^T + bias ---------------
// A: [M x K] bf16 (lda), W: [N x K] bf16 (ldb), C: [M x N] f32 (ldc).
// One wave -> 32(M) x 64(N) tile.  M%32==0, N%64==0, K%32==0 required.

__global__ __launch_bounds__(256)
void gemm_bf16_wmma(const bf16* __restrict__ A, const bf16* __restrict__ W,
                    const float* __restrict__ bias, float* __restrict__ C,
                    int M, int N, int K, int lda, int ldb, int ldc) {
    int wave   = blockIdx.x * (blockDim.x >> 5) + (threadIdx.x >> 5);
    int tilesN = N >> 6;
    int tilesM = M >> 5;
    if (wave >= tilesM * tilesN) return;           // uniform per wave: EXEC all-1s
    int m0   = (wave / tilesN) << 5;
    int n0   = (wave % tilesN) << 6;
    int lane = threadIdx.x & 31;
    int half = lane >> 4;
    int lr   = lane & 15;

    floatx8 acc[2][4];
#pragma unroll
    for (int i = 0; i < 2; i++)
#pragma unroll
        for (int j = 0; j < 4; j++)
            acc[i][j] = (floatx8){0.f,0.f,0.f,0.f,0.f,0.f,0.f,0.f};

    for (int k0 = 0; k0 < K; k0 += 32) {
        bf16x16 a0 = load_frag_a(A, lda, m0 + lr,      k0, half);
        bf16x16 a1 = load_frag_a(A, lda, m0 + 16 + lr, k0, half);
#pragma unroll
        for (int j = 0; j < 4; j++) {
            bf16x16 b = load_frag_b(W, ldb, n0 + j * 16 + lr, k0, half);
            acc[0][j] = __builtin_amdgcn_wmma_f32_16x16x32_bf16(
                false, a0, false, b, (short)0, acc[0][j], false, false);
            acc[1][j] = __builtin_amdgcn_wmma_f32_16x16x32_bf16(
                false, a1, false, b, (short)0, acc[1][j], false, false);
        }
    }

    // C/D layout: VGPR r -> M = m0 + 16*i + 8*half + r ; N = n0 + 16*j + lr
#pragma unroll
    for (int j = 0; j < 4; j++) {
        int n = n0 + j * 16 + lr;
        float bv = bias[n];
#pragma unroll
        for (int i = 0; i < 2; i++) {
            int mb = m0 + i * 16 + half * 8;
#pragma unroll
            for (int r = 0; r < 8; r++)
                C[(size_t)(mb + r) * ldc + n] = acc[i][j][r] + bv;
        }
    }
}

// ---------------- elementwise / epilogue kernels ----------------------------

__global__ void f32_to_bf16_k(bf16* __restrict__ dst, const float* __restrict__ src, int n) {
    int i = blockIdx.x * blockDim.x + threadIdx.x;
    if (i < n) dst[i] = (bf16)src[i];
}

__global__ void conv_pad_k(bf16* __restrict__ dst, const float* __restrict__ src,
                           int rows, int scols, int dcols) {
    int i = blockIdx.x * blockDim.x + threadIdx.x;
    if (i >= rows * dcols) return;
    int r = i / dcols, c = i - r * dcols;
    dst[i] = (c < scols) ? (bf16)src[(size_t)r * scols + c] : (bf16)0.0f;
}

__global__ void fill_actions_k(bf16* __restrict__ xin, const float* __restrict__ act) {
    int i = blockIdx.x * blockDim.x + threadIdx.x;   // 512*3
    if (i >= BB * 3) return;
    int b = i / 3, j = i - b * 3;
    xin[(size_t)b * KPRE + SF + j] = (bf16)act[i];
}

__global__ __launch_bounds__(256)
void ln_silu_k(const float* __restrict__ in, bf16* __restrict__ out,
               const float* __restrict__ g, const float* __restrict__ beta, int ncols) {
    int row = blockIdx.x;
    int tid = threadIdx.x;
    const float* x = in + (size_t)row * ncols;
    float s = 0.f, s2 = 0.f;
    for (int c = tid; c < ncols; c += blockDim.x) { float v = x[c]; s += v; s2 += v * v; }
    __shared__ float sh[512];
    sh[tid] = s; sh[256 + tid] = s2;
    __syncthreads();
    for (int off = 128; off > 0; off >>= 1) {
        if (tid < off) { sh[tid] += sh[tid + off]; sh[256 + tid] += sh[256 + tid + off]; }
        __syncthreads();
    }
    float mean = sh[0] / ncols;
    float var  = sh[256] / ncols - mean * mean;
    float rstd = rsqrtf(var + 1e-5f);
    for (int c = tid; c < ncols; c += blockDim.x) {
        float v = (x[c] - mean) * rstd * g[c] + beta[c];
        float sig = 1.f / (1.f + __expf(-v));
        out[(size_t)row * ncols + c] = (bf16)(v * sig);
    }
}

__global__ void gru_gates_k(const float* __restrict__ gi, const float* __restrict__ gh,
                            float* __restrict__ deter, bf16* __restrict__ deter_bf,
                            float* __restrict__ out_h /* t-slice base */) {
    int i = blockIdx.x * blockDim.x + threadIdx.x;   // 512*512
    if (i >= BB * DET) return;
    int b = i / DET, n = i - b * DET;
    size_t gbase = (size_t)b * (3 * DET);
    float ir = gi[gbase + n],           hr = gh[gbase + n];
    float iz = gi[gbase + DET + n],     hz = gh[gbase + DET + n];
    float in_ = gi[gbase + 2*DET + n],  hn = gh[gbase + 2*DET + n];
    float r = 1.f / (1.f + __expf(-(ir + hr)));
    float z = 1.f / (1.f + __expf(-(iz + hz)));
    float nn = tanhf(in_ + r * hn);
    float h = (1.f - z) * nn + z * deter[i];
    deter[i]    = h;
    deter_bf[i] = (bf16)h;
    out_h[(size_t)b * OUTW + n] = h;
}

__global__ void build_qin_k(bf16* __restrict__ qin, const float* __restrict__ h,
                            const float* __restrict__ emb /* t-slice */) {
    int i = blockIdx.x * blockDim.x + threadIdx.x;   // 512*1024
    if (i >= BB * (DET + EMB)) return;
    int b = i / (DET + EMB), c = i - b * (DET + EMB);
    float v = (c < DET) ? h[(size_t)b * DET + c] : emb[(size_t)b * EMB + (c - DET)];
    qin[i] = (bf16)v;
}

__device__ __forceinline__ float hash_uniform(unsigned int a, unsigned int b, unsigned int c) {
    unsigned int s = a * 0x9E3779B9u ^ b * 0x85EBCA6Bu ^ c * 0xC2B2AE35u;
    s ^= s >> 16; s *= 0x7FEB352Du;
    s ^= s >> 15; s *= 0x846CA68Bu;
    s ^= s >> 16;
    return (float)(s >> 8) * (1.0f / 16777216.0f);
}

__global__ void sample_k(float* __restrict__ out_t /* t-slice base */,
                         bf16* __restrict__ xin, int t) {
    int gid = blockIdx.x * blockDim.x + threadIdx.x;  // 512*32 (b, stoch-group)
    if (gid >= BB * 32) return;
    int b = gid >> 5, g = gid & 31;
    float* row = out_t + (size_t)b * OUTW;
    const float* lg = row + DET + SF + SF + g * 32;   // post_logits slot
    float m = -3.4e38f;
    float p[32];
#pragma unroll
    for (int c = 0; c < 32; c++) { p[c] = lg[c]; m = fmaxf(m, p[c]); }
    float s = 0.f;
#pragma unroll
    for (int c = 0; c < 32; c++) { p[c] = __expf(p[c] - m); s += p[c]; }
    float inv = 1.f / s;
#pragma unroll
    for (int c = 0; c < 32; c++) p[c] = 0.99f * p[c] * inv + 0.01f / 32.f;
    float u = hash_uniform((unsigned)(2 * t + 1), (unsigned)b, (unsigned)g);
    int idx = 31;
    float cdf = 0.f;
#pragma unroll
    for (int c = 0; c < 32; c++) { cdf += p[c]; if (u < cdf) { idx = c; break; } }
#pragma unroll
    for (int c = 0; c < 32; c++) {
        float v = (c == idx) ? 1.f : 0.f;
        row[DET + g * 32 + c] = v;                       // post_stoch (one-hot ST sample)
        xin[(size_t)b * KPRE + g * 32 + c] = (bf16)v;    // feeds next step's pre-GRU
    }
}

// ---------------------------------------------------------------------------

static inline int cdiv_i(int a, int b) { return (a + b - 1) / b; }

static void launch_gemm(const bf16* A, const bf16* W, const float* bias, float* C,
                        int M, int N, int K, int lda, int ldb, int ldc, hipStream_t s) {
    int waves = (M / 32) * (N / 64);
    gemm_bf16_wmma<<<cdiv_i(waves, 8), 256, 0, s>>>(A, W, bias, C, M, N, K, lda, ldb, ldc);
}

extern "C" void kernel_launch(void* const* d_in, const int* in_sizes, int n_in,
                              void* d_out, int out_size, void* d_ws, size_t ws_size,
                              hipStream_t stream) {
    const float* actions    = (const float*)d_in[0];
    const float* embed      = (const float*)d_in[1];
    const float* pre_w      = (const float*)d_in[2];
    const float* pre_b      = (const float*)d_in[3];
    const float* pre_g      = (const float*)d_in[4];
    const float* pre_beta   = (const float*)d_in[5];
    const float* gru_wih    = (const float*)d_in[6];
    const float* gru_whh    = (const float*)d_in[7];
    const float* gru_bih    = (const float*)d_in[8];
    const float* gru_bhh    = (const float*)d_in[9];
    const float* prior_w1   = (const float*)d_in[10];
    const float* prior_b1   = (const float*)d_in[11];
    const float* prior_g    = (const float*)d_in[12];
    const float* prior_beta = (const float*)d_in[13];
    const float* prior_w2   = (const float*)d_in[14];
    const float* prior_b2   = (const float*)d_in[15];
    const float* post_w1    = (const float*)d_in[16];
    const float* post_b1    = (const float*)d_in[17];
    const float* post_g     = (const float*)d_in[18];
    const float* post_beta  = (const float*)d_in[19];
    const float* post_w2    = (const float*)d_in[20];
    const float* post_b2    = (const float*)d_in[21];
    float* out = (float*)d_out;

    // ---- workspace layout (256B aligned) ----
    char* ws = (char*)d_ws;
    size_t off = 0;
    auto alloc = [&](size_t bytes) {
        void* p = ws + off;
        off = (off + bytes + 255) & ~(size_t)255;
        return p;
    };
    bf16*  w_pre  = (bf16*)alloc((size_t)HID * KPRE * 2);
    bf16*  w_ih   = (bf16*)alloc((size_t)3 * DET * HID * 2);
    bf16*  w_hh   = (bf16*)alloc((size_t)3 * DET * DET * 2);
    bf16*  w_p1   = (bf16*)alloc((size_t)HID * DET * 2);
    bf16*  w_p2   = (bf16*)alloc((size_t)SF  * HID * 2);
    bf16*  w_q1   = (bf16*)alloc((size_t)HID * (DET + EMB) * 2);
    bf16*  w_q2   = (bf16*)alloc((size_t)SF  * HID * 2);
    bf16*  xin    = (bf16*)alloc((size_t)BB * KPRE * 2);
    float* preraw = (float*)alloc((size_t)BB * HID * 4);
    bf16*  x_bf   = (bf16*)alloc((size_t)BB * HID * 2);
    float* gi     = (float*)alloc((size_t)BB * 3 * DET * 4);
    float* gh     = (float*)alloc((size_t)BB * 3 * DET * 4);
    float* deterf = (float*)alloc((size_t)BB * DET * 4);
    bf16*  deterb = (bf16*)alloc((size_t)BB * DET * 2);
    float* tmp1   = (float*)alloc((size_t)BB * HID * 4);
    bf16*  h1_bf  = (bf16*)alloc((size_t)BB * HID * 2);
    bf16*  qin    = (bf16*)alloc((size_t)BB * (DET + EMB) * 2);
    (void)ws_size; (void)in_sizes; (void)n_in; (void)out_size;

    // ---- weights -> bf16 (deterministic every call; ~8 MB, trivial) ----
    auto conv = [&](bf16* dst, const float* src, int n) {
        f32_to_bf16_k<<<cdiv_i(n, 256), 256, 0, stream>>>(dst, src, n);
    };
    conv_pad_k<<<cdiv_i(HID * KPRE, 256), 256, 0, stream>>>(w_pre, pre_w, HID, SF + 3, KPRE);
    conv(w_ih, gru_wih, 3 * DET * HID);
    conv(w_hh, gru_whh, 3 * DET * DET);
    conv(w_p1, prior_w1, HID * DET);
    conv(w_p2, prior_w2, SF * HID);
    conv(w_q1, post_w1,  HID * (DET + EMB));
    conv(w_q2, post_w2,  SF * HID);

    // ---- initial state: stoch = 0, deter = 0, pad cols of xin = 0 ----
    hipMemsetAsync(xin,    0, (size_t)BB * KPRE * 2, stream);
    hipMemsetAsync(deterf, 0, (size_t)BB * DET * 4, stream);
    hipMemsetAsync(deterb, 0, (size_t)BB * DET * 2, stream);

    for (int t = 0; t < TT; t++) {
        float* out_t = out + (size_t)t * BB * OUTW;

        // xin[:, SF:SF+3] = action_t
        fill_actions_k<<<cdiv_i(BB * 3, 256), 256, 0, stream>>>(
            xin, actions + (size_t)t * BB * 3);

        // pre-GRU linear + LN + SiLU
        launch_gemm(xin, w_pre, pre_b, preraw, BB, HID, KPRE, KPRE, KPRE, HID, stream);
        ln_silu_k<<<BB, 256, 0, stream>>>(preraw, x_bf, pre_g, pre_beta, HID);

        // GRU
        launch_gemm(x_bf,   w_ih, gru_bih, gi, BB, 3 * DET, HID, HID, HID, 3 * DET, stream);
        launch_gemm(deterb, w_hh, gru_bhh, gh, BB, 3 * DET, DET, DET, DET, 3 * DET, stream);
        gru_gates_k<<<cdiv_i(BB * DET, 256), 256, 0, stream>>>(gi, gh, deterf, deterb, out_t);

        // prior net -> prior_logits written straight into output slice
        launch_gemm(deterb, w_p1, prior_b1, tmp1, BB, HID, DET, DET, DET, HID, stream);
        ln_silu_k<<<BB, 256, 0, stream>>>(tmp1, h1_bf, prior_g, prior_beta, HID);
        launch_gemm(h1_bf, w_p2, prior_b2, out_t + DET + SF,
                    BB, SF, HID, HID, HID, OUTW, stream);

        // posterior net -> post_logits written straight into output slice
        build_qin_k<<<cdiv_i(BB * (DET + EMB), 256), 256, 0, stream>>>(
            qin, deterf, embed + (size_t)t * BB * EMB);
        launch_gemm(qin, w_q1, post_b1, tmp1, BB, HID, DET + EMB,
                    DET + EMB, DET + EMB, HID, stream);
        ln_silu_k<<<BB, 256, 0, stream>>>(tmp1, h1_bf, post_g, post_beta, HID);
        launch_gemm(h1_bf, w_q2, post_b2, out_t + DET + SF + SF,
                    BB, SF, HID, HID, HID, OUTW, stream);

        // unimix softmax + categorical one-hot (ST sample); feeds next xin
        sample_k<<<cdiv_i(BB * 32, 256), 256, 0, stream>>>(out_t, xin, t);
    }
}
